// Net_36524401885621
// MI455X (gfx1250) — matrix operand order repaired and checked by reading, not compiled, not run on hardware
//
#include <hip/hip_runtime.h>
#include <cstdint>

typedef __attribute__((ext_vector_type(16))) _Float16 v16h;
typedef __attribute__((ext_vector_type(8)))  float    v8f;

#define SEQ   2048
#define EMB   300
#define NC1   32
#define NC2   128
#define KW    5
#define P1    2047          // valid pool1 positions (maxpool(300) of 614396)
#define P1PAD 2052          // padded so conv2 tile reads stay in-bounds (zeros)
#define T2    2043          // conv2 output positions
#define NTILE 128           // ceil(2043/16) position tiles

// ---------------------------------------------------------------------------
// Kernel 1: w2 [c2][c1][k] f32  ->  w2h [k][c2][c1] f16  (B-matrix friendly)
// ---------------------------------------------------------------------------
__global__ __launch_bounds__(256)
void w2_convert_kernel(const float* __restrict__ w2, _Float16* __restrict__ w2h) {
    int i = blockIdx.x * 256 + threadIdx.x;
    if (i >= KW * NC2 * NC1) return;
    int c1 = i & 31;
    int c2 = (i >> 5) & 127;
    int kk = i >> 12;               // 128*32 = 4096 per k-slice
    w2h[i] = (_Float16)w2[(c2 * NC1 + c1) * KW + kk];
}

// ---------------------------------------------------------------------------
// Kernel 2: fused embedding-gather + conv1(Cin=1) + ReLU + maxpool(300).
// Pool windows align exactly with sequence rows (stride 300 == EMB).
// Block = (8 windows of one batch); wave = one window; lane = channel (32!).
// Output pool1h [b][t'][c1] f16, rows t' in [0,2052), t' >= 2047 zeroed.
// ---------------------------------------------------------------------------
__global__ __launch_bounds__(256)
void conv1_pool_kernel(const int* __restrict__ x, const float* __restrict__ emb,
                       const float* __restrict__ w1, const float* __restrict__ b1,
                       _Float16* __restrict__ pool1h)
{
    __shared__ float lds[8 * EMB + KW];           // 2405 floats
    const int b  = blockIdx.y;
    const int p0 = blockIdx.x * 8;

    // Stage 8 embedding rows + 5 floats of the next row into LDS.
    for (int i = threadIdx.x; i < 8 * EMB + KW; i += 256) {
        int row = i / EMB;
        int d   = i - row * EMB;
        int s   = p0 + row;
        float v = 0.0f;
        if (s < SEQ) v = emb[(long)x[b * SEQ + s] * EMB + d];
        lds[i] = v;
    }
    __syncthreads();

    const int w = threadIdx.x >> 5;               // window within block
    const int c = threadIdx.x & 31;               // channel == lane
    const int p = p0 + w;

    const float k0 = w1[c * KW + 0], k1 = w1[c * KW + 1], k2 = w1[c * KW + 2],
                k3 = w1[c * KW + 3], k4 = w1[c * KW + 4];
    const float bias = b1[c];

    const float* e = &lds[w * EMB];
    float e0 = e[0], e1 = e[1], e2 = e[2], e3 = e[3];
    float mx = 0.0f;                              // relu >= 0, so 0 is identity
    #pragma unroll 4
    for (int j = 0; j < EMB; ++j) {
        float e4  = e[j + 4];
        float acc = fmaf(e0, k0, fmaf(e1, k1, fmaf(e2, k2,
                    fmaf(e3, k3, fmaf(e4, k4, bias)))));
        acc = acc > 0.0f ? acc : 0.0f;
        mx  = fmaxf(mx, acc);
        e0 = e1; e1 = e2; e2 = e3; e3 = e4;
    }
    if (p < P1) pool1h[((long)b * P1PAD + p) * NC1 + c] = (_Float16)mx;

    // Zero the padding rows once per batch.
    if (blockIdx.x == 0 && threadIdx.x < (P1PAD - P1) * NC1) {
        int t = P1 + (threadIdx.x >> 5);
        pool1h[((long)b * P1PAD + t) * NC1 + (threadIdx.x & 31)] = (_Float16)0.0f;
    }
}

// ---------------------------------------------------------------------------
// Kernel 3: conv2 as WMMA GEMM, K = 160 = 5 ksteps x 32(c1).
// Block = (batch, 16-position tile); wave = 16-output-channel N-tile.
// A (16x32 f16, ISA layout): lane row = t0+laneN+kk, K = {0..7,16..23} or
//   {8..15,24..31} -> two 16B chunks of the 64B pool1h row.
// B (32x16 f16): lane col = n0+laneN, K = same chunks as A (table order).
// C layout: VGPR j, lanes 0-15 -> M=j, lanes 16-31 -> M=j+8.
// Fused bias + ReLU + masked max over tile, lane-pair merge via shfl_xor(16).
// ---------------------------------------------------------------------------
__global__ __launch_bounds__(256)
void conv2_wmma_kernel(const _Float16* __restrict__ pool1h,
                       const _Float16* __restrict__ w2h,
                       const float* __restrict__ b2,
                       float* __restrict__ tilemax)
{
    const int b      = blockIdx.y;
    const int tile   = blockIdx.x;
    const int t0     = tile * 16;
    const int w      = threadIdx.x >> 5;
    const int lane   = threadIdx.x & 31;
    const int laneN  = lane & 15;
    const int laneHi = lane >> 4;
    const int n0     = w * 16;

    union V16 { v16h h; uint32_t u[8]; };
    v8f acc = {};

    #pragma unroll
    for (int kk = 0; kk < KW; ++kk) {
        V16 a, bm;
        const uint32_t* rowA =
            (const uint32_t*)(pool1h + ((long)b * P1PAD + t0 + laneN + kk) * NC1);
        const int ba = laneHi * 4;                // dword offset of K-chunk
        #pragma unroll
        for (int q = 0; q < 4; ++q) {
            a.u[q]     = rowA[ba + q];            // K = base .. base+7
            a.u[4 + q] = rowA[8 + ba + q];        // K = base+16 .. base+23
        }
        const uint32_t* rowB =
            (const uint32_t*)(w2h + ((kk * NC2) + n0 + laneN) * NC1);
        #pragma unroll
        for (int q = 0; q < 8; ++q) bm.u[q] = rowB[laneHi * 4 + (q & 3) + (q >> 2) * 8];

        acc = __builtin_amdgcn_wmma_f32_16x16x32_f16(
                  false, a.h, false, bm.h, (short)0, acc, false, false);
    }

    const float bias = b2[n0 + laneN];
    float m = 0.0f;
    #pragma unroll
    for (int j = 0; j < 8; ++j) {
        int   t = t0 + j + laneHi * 8;
        float v = acc[j] + bias;
        v = v > 0.0f ? v : 0.0f;                  // ReLU
        if (t < T2) m = fmaxf(m, v);              // mask tail positions
    }
    m = fmaxf(m, __shfl_xor(m, 16, 32));          // merge M=0..7 with M=8..15
    if (laneHi == 0)
        tilemax[(((long)b * NTILE) + tile) * NC2 + n0 + laneN] = m;
}

// ---------------------------------------------------------------------------
// Kernel 4: reduce tile maxima, 128->5 linear, softmax. One block.
// ---------------------------------------------------------------------------
__global__ __launch_bounds__(256)
void head_kernel(const float* __restrict__ tilemax,
                 const float* __restrict__ wl, const float* __restrict__ bl,
                 float* __restrict__ out)
{
    __shared__ float pooled[8 * NC2];
    __shared__ float logits[8 * 5];

    for (int i = threadIdx.x; i < 8 * NC2; i += 256) {
        int b = i >> 7, c = i & 127;
        float m = 0.0f;
        const float* p = tilemax + ((long)b * NTILE) * NC2 + c;
        for (int t = 0; t < NTILE; ++t) m = fmaxf(m, p[t * NC2]);
        pooled[i] = m;
    }
    __syncthreads();

    if (threadIdx.x < 8 * 5) {
        int b = threadIdx.x / 5, n = threadIdx.x % 5;
        float a = bl[n];
        for (int c = 0; c < NC2; ++c) a += pooled[b * NC2 + c] * wl[n * NC2 + c];
        logits[threadIdx.x] = a;
    }
    __syncthreads();

    if (threadIdx.x < 8) {
        int b = threadIdx.x;
        float mx = logits[b * 5];
        for (int n = 1; n < 5; ++n) mx = fmaxf(mx, logits[b * 5 + n]);
        float e[5], s = 0.0f;
        for (int n = 0; n < 5; ++n) { e[n] = expf(logits[b * 5 + n] - mx); s += e[n]; }
        for (int n = 0; n < 5; ++n) out[b * 5 + n] = e[n] / s;
    }
}

// ---------------------------------------------------------------------------
extern "C" void kernel_launch(void* const* d_in, const int* in_sizes, int n_in,
                              void* d_out, int out_size, void* d_ws, size_t ws_size,
                              hipStream_t stream)
{
    const int*   x   = (const int*)  d_in[0];
    const float* emb = (const float*)d_in[1];
    const float* w1  = (const float*)d_in[2];
    const float* b1  = (const float*)d_in[3];
    const float* w2  = (const float*)d_in[4];
    const float* b2  = (const float*)d_in[5];
    const float* wl  = (const float*)d_in[6];
    const float* bl  = (const float*)d_in[7];
    float* out = (float*)d_out;

    // Workspace layout (all 64B-aligned offsets):
    //   w2h:     5*128*32 f16            = 40,960 B
    //   pool1h:  8*2052*32 f16           = 1,050,624 B
    //   tilemax: 8*128*128 f32           = 524,288 B      (total ~1.6 MB)
    char* ws = (char*)d_ws;
    _Float16* w2h     = (_Float16*)(ws);
    _Float16* pool1h  = (_Float16*)(ws + 40960);
    float*    tilemax = (float*)   (ws + 40960 + (long)8 * P1PAD * NC1 * 2);

    hipLaunchKernelGGL(w2_convert_kernel, dim3((KW*NC2*NC1 + 255) / 256), dim3(256),
                       0, stream, w2, w2h);
    hipLaunchKernelGGL(conv1_pool_kernel, dim3(256, 8), dim3(256),
                       0, stream, x, emb, w1, b1, pool1h);
    hipLaunchKernelGGL(conv2_wmma_kernel, dim3(NTILE, 8), dim3(256),
                       0, stream, pool1h, w2h, b2, tilemax);
    hipLaunchKernelGGL(head_kernel, dim3(1), dim3(256),
                       0, stream, tilemax, wl, bl, out);
}
